// QuantumAttentionMechanism_51599737094580
// MI455X (gfx1250) — matrix operand order repaired and checked by reading, not compiled, not run on hardware
//
#include <hip/hip_runtime.h>
#include <hip/hip_bf16.h>

// ---------------- problem constants ----------------
constexpr int Bb   = 4;
constexpr int Ss   = 1024;
constexpr int Dm   = 1024;
constexpr int NQ   = 8;
constexpr float ALPHA = 0.7f;
constexpr float BETA  = 0.3f;   // 1 - ALPHA
// mean over H heads of (Qh.Kh^T)/sqrt(64) == (Q.K^T)/128

// ---------------- WMMA types ----------------
typedef __attribute__((ext_vector_type(16))) __bf16 v16bf;
typedef __attribute__((ext_vector_type(8)))  float  v8f;

union Frag { v16bf v; unsigned int u[8]; };

__device__ __forceinline__ unsigned short f32_to_bf16_bits(float f) {
  __bf16 h = (__bf16)f;
  return __builtin_bit_cast(unsigned short, h);
}

// ---------------- async copy (CDNA5 GLOBAL_LOAD_ASYNC_TO_LDS, GVS mode) ----
// LDS byte address from a VGPR; global address = SGPR base + per-lane VGPR byte offset
__device__ __forceinline__ void async_b128(unsigned lds_off,
                                           const unsigned short* sbase,
                                           unsigned gbyte_off) {
  asm volatile("global_load_async_to_lds_b128 %0, %1, %2"
               :
               : "v"(lds_off), "v"(gbyte_off), "s"(sbase)
               : "memory");
}
__device__ __forceinline__ void wait_async_le6() {
  asm volatile("s_wait_asynccnt 0x6" ::: "memory");
}
__device__ __forceinline__ void wait_async_0() {
  asm volatile("s_wait_asynccnt 0x0" ::: "memory");
}

// ---------------- bf16 WMMA GEMM (all operands bf16, B always transposed) --
// C[m,n] = scale * sum_k A[m,k]*B[n,k]  (+bias[n]);  OUT_T stores C[n][m].
// Block tile 128x64, BK=64, 256 threads = 8 waves; wave tile 32x32 (4 WMMAs
// per K-substep, 8 per staged tile). Double-buffered LDS fed by async copies.
constexpr int BMt = 128, BNt = 64, BKt = 64;
constexpr int PITCH = 72;   // ushort pitch: 144B rows (16B aligned segments)

__device__ __forceinline__ void stage_tiles(unsigned short* AsBuf,
                                            unsigned short* BsBuf,
                                            const unsigned short* Ab,
                                            const unsigned short* Bp,
                                            int bm, int bn, int lda, int ldb,
                                            int k0, int tid) {
  // A tile: 128 rows x 64 k = 1024 x 16B segments -> 4 per thread
#pragma unroll
  for (int i = 0; i < 4; ++i) {
    const int s = tid + 256 * i;
    const int row = s >> 3, seg = s & 7;
    const unsigned g = (unsigned)(((bm + row) * lda + k0 + seg * 8) * 2);
    const unsigned l = (unsigned)(unsigned long long)(&AsBuf[row * PITCH + seg * 8]);
    async_b128(l, Ab, g);
  }
  // B tile: 64 rows(n) x 64 k = 512 x 16B segments -> 2 per thread
#pragma unroll
  for (int i = 0; i < 2; ++i) {
    const int s = tid + 256 * i;
    const int n = s >> 3, seg = s & 7;
    const unsigned g = (unsigned)(((bn + n) * ldb + k0 + seg * 8) * 2);
    const unsigned l = (unsigned)(unsigned long long)(&BsBuf[n * PITCH + seg * 8]);
    async_b128(l, Bp, g);
  }
}

template<bool OUT_F32, bool BIAS, bool OUT_T>
__global__ __launch_bounds__(256)
void gemm_bf16_wmma_async(const unsigned short* __restrict__ A,
                          const unsigned short* __restrict__ B,
                          const float* __restrict__ bias, void* __restrict__ Cp,
                          int lda, int ldb, int ldc, int K,
                          long long strideA, long long strideB, long long strideC,
                          float scale)
{
  __shared__ __align__(16) unsigned short As[2][BMt * PITCH];
  __shared__ __align__(16) unsigned short Bs[2][BNt * PITCH];

  const int tid  = threadIdx.x;
  const int wave = tid >> 5;
  const int lane = tid & 31;
  const int hlf  = lane >> 4;
  const int l16  = lane & 15;
  const int wm   = wave & 3;    // M 32-row group (4 -> 128)
  const int wn   = wave >> 2;   // N 32-col group (2 -> 64)

  const int bm = blockIdx.y * BMt;
  const int bn = blockIdx.x * BNt;
  const unsigned short* Ab = A + (long long)blockIdx.z * strideA;
  const unsigned short* Bp = B + (long long)blockIdx.z * strideB;
  const long long cBase = (long long)blockIdx.z * strideC;

  v8f acc[2][2] = {{{}, {}}, {{}, {}}};

  const int niter = K / BKt;
  stage_tiles(As[0], Bs[0], Ab, Bp, bm, bn, lda, ldb, 0, tid);

  for (int it = 0; it < niter; ++it) {
    const int cur = it & 1;
    if (it + 1 < niter) {
      stage_tiles(As[cur ^ 1], Bs[cur ^ 1], Ab, Bp, bm, bn, lda, ldb,
                  (it + 1) * BKt, tid);
      wait_async_le6();   // current tile done; next tile (6 ops) in flight
    } else {
      wait_async_0();
    }
    __syncthreads();

#pragma unroll
    for (int ks = 0; ks < BKt; ks += 32) {
      Frag a0, a1, b0, b1;
      // A 16x32 bf16 frag (ISA 7.12.2): lanes = rows; VGPR p<4: K=hlf*8+2p;
      // p>=4: K=16+hlf*8+2(p-4)
      const int am0 = (wm * 32 + l16) * PITCH + ks;
      const int am1 = am0 + 16 * PITCH;
#pragma unroll
      for (int p = 0; p < 8; ++p) {
        const int kb = (p < 4) ? (hlf * 8 + 2 * p) : (16 + hlf * 8 + 2 * (p - 4));
        a0.u[p] = *(const unsigned int*)&As[cur][am0 + kb];
        a1.u[p] = *(const unsigned int*)&As[cur][am1 + kb];
      }
      // B 32x16 frag: lane -> col n=l16, VGPR p<4: K=hlf*8+2p; p>=4: K=16+hlf*8+2(p-4)
      const int bo0 = (wn * 32 + l16) * PITCH + ks;
      const int bo1 = bo0 + 16 * PITCH;
#pragma unroll
      for (int p = 0; p < 8; ++p) {
        const int kb = (p < 4) ? (hlf * 8 + 2 * p) : (16 + hlf * 8 + 2 * (p - 4));
        b0.u[p] = *(const unsigned int*)&Bs[cur][bo0 + kb];
        b1.u[p] = *(const unsigned int*)&Bs[cur][bo1 + kb];
      }
      acc[0][0] = __builtin_amdgcn_wmma_f32_16x16x32_bf16(false, a0.v, false, b0.v, (short)0, acc[0][0], false, false);
      acc[0][1] = __builtin_amdgcn_wmma_f32_16x16x32_bf16(false, a0.v, false, b1.v, (short)0, acc[0][1], false, false);
      acc[1][0] = __builtin_amdgcn_wmma_f32_16x16x32_bf16(false, a1.v, false, b0.v, (short)0, acc[1][0], false, false);
      acc[1][1] = __builtin_amdgcn_wmma_f32_16x16x32_bf16(false, a1.v, false, b1.v, (short)0, acc[1][1], false, false);
    }
    __syncthreads();
  }

  // epilogue: per 16x16 tile, VGPR r -> M = r + 8*hlf, N = l16
#pragma unroll
  for (int mi = 0; mi < 2; ++mi) {
#pragma unroll
    for (int ni = 0; ni < 2; ++ni) {
#pragma unroll
      for (int r = 0; r < 8; ++r) {
        const int m = bm + wm * 32 + mi * 16 + hlf * 8 + r;
        const int n = bn + wn * 32 + ni * 16 + l16;
        float v = acc[mi][ni][r] * scale;
        if (BIAS) v += bias[n];
        const long long off = OUT_T ? (cBase + (long long)n * ldc + m)
                                    : (cBase + (long long)m * ldc + n);
        if (OUT_F32) ((float*)Cp)[off] = v;
        else         ((unsigned short*)Cp)[off] = f32_to_bf16_bits(v);
      }
    }
  }
}

// ---------------- fp32 -> bf16 bulk convert ----------------
__global__ __launch_bounds__(256)
void cvt_f32_bf16(const float* __restrict__ s, unsigned short* __restrict__ d,
                  int n4) {
  const int i = blockIdx.x * 256 + threadIdx.x;
  if (i >= n4) return;
  const float4 v = reinterpret_cast<const float4*>(s)[i];
  unsigned short r[4] = {f32_to_bf16_bits(v.x), f32_to_bf16_bits(v.y),
                         f32_to_bf16_bits(v.z), f32_to_bf16_bits(v.w)};
  reinterpret_cast<uint2*>(d)[i] = *reinterpret_cast<uint2*>(r);
}

// ---------------- entropy row statistics ----------------
__global__ __launch_bounds__(256)
void entropy_stats_kernel(const float* __restrict__ q, const float* __restrict__ k,
                          float* __restrict__ Eq, float* __restrict__ Aq,
                          float* __restrict__ Ek, float* __restrict__ Ak, int rows)
{
  const int i = blockIdx.x * blockDim.x + threadIdx.x;
  if (i >= 2 * rows) return;
  const bool isQ = (i < rows);
  const long long base = (long long)(isQ ? i : (i - rows)) * Dm;
  const float* src = isQ ? (q + base) : (k + base);
  float E = 0.f, A = 0.f;
#pragma unroll
  for (int m = 0; m < NQ; ++m) {
    float t = tanhf(src[m]);
    float e = expf(t);
    E += e;
    A += e * t;
  }
  if (isQ) { Eq[i] = E; Aq[i] = A; }
  else     { Ek[i - rows] = E; Ak[i - rows] = A; }
}

// ---------------- fused combine + softmax ----------------
// scores hold ALPHA*(Q.K^T)/128.  entropy(i,j)=ln(Eq+Ek)-(Aq+Ak)/(Eq+Ek)
__global__ __launch_bounds__(256)
void combine_softmax_kernel(const float* __restrict__ scores,
                            const float* __restrict__ Eq, const float* __restrict__ Aq,
                            const float* __restrict__ Ek, const float* __restrict__ Ak,
                            float* __restrict__ attn_f32,
                            unsigned short* __restrict__ attn_bf)
{
  __shared__ float red[256];
  const int row = blockIdx.x;
  const int b   = row >> 10;
  const long long base  = (long long)row * Ss;
  const long long ebase = (long long)b * Ss;
  const float eqi = Eq[row];
  const float aqi = Aq[row];

  float c[4];
  float mx = -1e30f;
#pragma unroll
  for (int p = 0; p < 4; ++p) {
    const int j = threadIdx.x + p * 256;
    const float denom = eqi + Ek[ebase + j];
    const float ent   = logf(denom) - (aqi + Ak[ebase + j]) / denom;
    const float v     = scores[base + j] + BETA * ent;
    c[p] = v;
    mx = fmaxf(mx, v);
  }
  red[threadIdx.x] = mx; __syncthreads();
  for (int s = 128; s > 0; s >>= 1) {
    if (threadIdx.x < s) red[threadIdx.x] = fmaxf(red[threadIdx.x], red[threadIdx.x + s]);
    __syncthreads();
  }
  mx = red[0]; __syncthreads();

  float sum = 0.f;
#pragma unroll
  for (int p = 0; p < 4; ++p) { c[p] = expf(c[p] - mx); sum += c[p]; }
  red[threadIdx.x] = sum; __syncthreads();
  for (int s = 128; s > 0; s >>= 1) {
    if (threadIdx.x < s) red[threadIdx.x] += red[threadIdx.x + s];
    __syncthreads();
  }
  const float inv = 1.f / red[0];

#pragma unroll
  for (int p = 0; p < 4; ++p) {
    const int j = threadIdx.x + p * 256;
    const float a = c[p] * inv;
    attn_f32[base + j] = a;
    attn_bf[base + j]  = f32_to_bf16_bits(a);
  }
}

// ---------------- host launcher ----------------
extern "C" void kernel_launch(void* const* d_in, const int* in_sizes, int n_in,
                              void* d_out, int out_size, void* d_ws, size_t ws_size,
                              hipStream_t stream) {
  const float* query = (const float*)d_in[0];
  const float* key   = (const float*)d_in[1];
  const float* value = (const float*)d_in[2];
  const float* Wq = (const float*)d_in[3];
  const float* bq = (const float*)d_in[4];
  const float* Wk = (const float*)d_in[5];
  const float* bk = (const float*)d_in[6];
  const float* Wv = (const float*)d_in[7];
  const float* bv = (const float*)d_in[8];
  const float* Wo = (const float*)d_in[9];
  const float* bo = (const float*)d_in[10];

  float* out_f32  = (float*)d_out;                       // [B,S,D]
  float* attn_out = out_f32 + (size_t)Bb * Ss * Dm;      // [B,S,S]

  const size_t nBSD = (size_t)Bb * Ss * Dm;   // 4M
  const size_t nBSS = (size_t)Bb * Ss * Ss;   // 4M
  const size_t nDD  = (size_t)Dm * Dm;        // 1M
  char* ws = (char*)d_ws;
  size_t off = 0;
  auto carve = [&](size_t bytes) { char* p = ws + off; off += bytes; return p; };
  unsigned short* qbf  = (unsigned short*)carve(nBSD * 2);
  unsigned short* kbf  = (unsigned short*)carve(nBSD * 2);
  unsigned short* vbf  = (unsigned short*)carve(nBSD * 2);
  unsigned short* Wqb  = (unsigned short*)carve(nDD * 2);
  unsigned short* Wkb  = (unsigned short*)carve(nDD * 2);
  unsigned short* Wvb  = (unsigned short*)carve(nDD * 2);
  unsigned short* Wob  = (unsigned short*)carve(nDD * 2);
  unsigned short* Qb   = (unsigned short*)carve(nBSD * 2);
  unsigned short* Kb   = (unsigned short*)carve(nBSD * 2);
  unsigned short* Vt   = (unsigned short*)carve(nBSD * 2);  // [B][D][S]
  unsigned short* ATTb = (unsigned short*)carve(nBSS * 2);
  unsigned short* ATDb = (unsigned short*)carve(nBSD * 2);
  float* scores = (float*)carve(nBSS * 4);
  float* Eq = (float*)carve((size_t)Bb * Ss * 4);
  float* Aq = (float*)carve((size_t)Bb * Ss * 4);
  float* Ek = (float*)carve((size_t)Bb * Ss * 4);
  float* Ak = (float*)carve((size_t)Bb * Ss * 4);

  const int rows = Bb * Ss;
  const long long sBSD = (long long)Ss * Dm;
  const long long sBSS = (long long)Ss * Ss;

  // 0) bulk fp32->bf16 conversion (one pass, amortized over 6 GEMMs)
  {
    const int n4a = (int)(nBSD / 4), n4w = (int)(nDD / 4);
    cvt_f32_bf16<<<(n4a + 255) / 256, 256, 0, stream>>>(query, qbf, n4a);
    cvt_f32_bf16<<<(n4a + 255) / 256, 256, 0, stream>>>(key,   kbf, n4a);
    cvt_f32_bf16<<<(n4a + 255) / 256, 256, 0, stream>>>(value, vbf, n4a);
    cvt_f32_bf16<<<(n4w + 255) / 256, 256, 0, stream>>>(Wq, Wqb, n4w);
    cvt_f32_bf16<<<(n4w + 255) / 256, 256, 0, stream>>>(Wk, Wkb, n4w);
    cvt_f32_bf16<<<(n4w + 255) / 256, 256, 0, stream>>>(Wv, Wvb, n4w);
    cvt_f32_bf16<<<(n4w + 255) / 256, 256, 0, stream>>>(Wo, Wob, n4w);
  }

  // 1) per-row entropy stats
  entropy_stats_kernel<<<(2 * rows + 255) / 256, 256, 0, stream>>>(
      query, key, Eq, Aq, Ek, Ak, rows);

  // 2) Q = q @ Wq^T + bq -> bf16 [B*S, D]
  dim3 gProj(Dm / BNt, (Bb * Ss) / BMt, 1);                 // 16 x 32
  gemm_bf16_wmma_async<false, true, false><<<gProj, 256, 0, stream>>>(
      qbf, Wqb, bq, Qb, Dm, Dm, Dm, Dm, 0, 0, 0, 1.0f);
  // 3) K
  gemm_bf16_wmma_async<false, true, false><<<gProj, 256, 0, stream>>>(
      kbf, Wkb, bk, Kb, Dm, Dm, Dm, Dm, 0, 0, 0, 1.0f);
  // 4) V, stored transposed per batch: Vt[b][d][t]
  dim3 gVproj(Dm / BNt, Ss / BMt, Bb);                      // 16 x 8 x 4
  gemm_bf16_wmma_async<false, true, true><<<gVproj, 256, 0, stream>>>(
      vbf, Wvb, bv, Vt, Dm, Dm, Ss, Dm, sBSD, 0, sBSD, 1.0f);

  // 5) scores = ALPHA/128 * Q @ K^T -> fp32 (batched)
  dim3 gScore(Ss / BNt, Ss / BMt, Bb);
  gemm_bf16_wmma_async<true, false, false><<<gScore, 256, 0, stream>>>(
      Qb, Kb, nullptr, scores, Dm, Dm, Ss, Dm, sBSD, sBSD, sBSS, ALPHA / 128.0f);

  // 6) combine with entropy + softmax -> attn (fp32 to d_out, bf16 to ws)
  combine_softmax_kernel<<<Bb * Ss, 256, 0, stream>>>(
      scores, Eq, Aq, Ek, Ak, attn_out, ATTb);

  // 7) attended = attn @ V -> bf16;  B(k=t,n=d) = Vt[d*S+t] (B_T layout)
  dim3 gAV(Dm / BNt, Ss / BMt, Bb);
  gemm_bf16_wmma_async<false, false, false><<<gAV, 256, 0, stream>>>(
      ATTb, Vt, nullptr, ATDb, Ss, Ss, Dm, Ss, sBSS, sBSD, sBSD, 1.0f);

  // 8) output = attended @ Wo^T + bo -> fp32 (d_out)
  gemm_bf16_wmma_async<true, true, false><<<gProj, 256, 0, stream>>>(
      ATDb, Wob, bo, out_f32, Dm, Dm, Dm, Dm, 0, 0, 0, 1.0f);
}
